// ConvEncoder_73933567033896
// MI455X (gfx1250) — matrix operand order repaired and checked by reading, not compiled, not run on hardware
//
#include <hip/hip_runtime.h>
#include <math.h>

#define BATCHN 16
#define SEQ    2048
#define EMBD   128
#define WINW   5
#define OUTD   128
#define KTOT   (EMBD * WINW)       // 640
#define MTILE  128                 // tokens per workgroup
#define EROWS  (MTILE + WINW - 1)  // 132 embedding rows staged
#define ESTR   136                 // halves per emb row in LDS (272B, bank-spread)
#define WSTR   40                  // halves per W-chunk row in LDS (80B, bank-spread)
#define WBUFH  (OUTD * WSTR)       // halves per W chunk buffer

typedef _Float16 v4h  __attribute__((ext_vector_type(4)));
typedef _Float16 v8h  __attribute__((ext_vector_type(8)));
typedef _Float16 v16h __attribute__((ext_vector_type(16)));
typedef float    v8f  __attribute__((ext_vector_type(8)));

// ---- CDNA5 async copy: global -> LDS, 16B per lane, tracked by ASYNCcnt ----
__device__ __forceinline__ void async_ld_b128(void* lds_dst, const void* gsrc) {
  unsigned off = (unsigned)(uintptr_t)lds_dst;  // low 32 bits of LDS aperture = byte offset
  asm volatile("global_load_async_to_lds_b128 %0, %1, off"
               :: "v"(off), "v"(gsrc)
               : "memory");
}

__device__ __forceinline__ void wait_async0() {
#if __has_builtin(__builtin_amdgcn_s_wait_asynccnt)
  __builtin_amdgcn_s_wait_asynccnt(0);
#else
  asm volatile("s_wait_asynccnt 0" ::: "memory");
#endif
}

// ---- prep: convert W (f32, row-major [OUTD][KTOT]) to f16 in workspace ----
__global__ void w_to_f16(const float* __restrict__ W, _Float16* __restrict__ Wh, int n) {
  int i = blockIdx.x * blockDim.x + threadIdx.x;
  if (i < n) Wh[i] = (_Float16)W[i];
}

// ---- main: fused gather + im2col GEMM (f16 WMMA, f32 accum) + bias + GELU ----
__global__ __launch_bounds__(256)
void conv_encoder_wmma(const int* __restrict__ x,
                       const float* __restrict__ emb,
                       const _Float16* __restrict__ Wh,
                       const float* __restrict__ bias,
                       float* __restrict__ out) {
  __shared__ int      idxLDS[EROWS];
  __shared__ _Float16 embLDS[EROWS * ESTR];  // ~35.9 KB
  __shared__ _Float16 wLDS[2 * WBUFH];       // 20 KB double-buffered W^T K-chunk

  const int tid  = threadIdx.x;
  const int wave = tid >> 5;    // 8 wave32s
  const int lane = tid & 31;
  const int ln   = lane & 15;   // N column / M row within subtile
  const int hi   = lane >> 4;   // lane-half selector
  const int mw   = wave & 3;    // wave's M-pair (2 subtiles = 32 tokens)
  const int nw   = wave >> 2;   // wave's N-half (4 subtiles = 64 outputs)

  const int blk = blockIdx.x;
  const int bb  = blk >> 4;              // batch row
  const int s0  = (blk & 15) * MTILE;    // first token of this tile

  const int wn  = tid >> 1;         // W-chunk staging: row n (0..127)
  const int wkp = (tid & 1) * 16;   // half-offset within 32-wide chunk

  // ---- kick off async staging of W chunk 0 immediately (hides behind gather)
  {
    const _Float16* src = Wh + (size_t)wn * KTOT + wkp;   // kb=0: w=0, e0=0
    async_ld_b128(wLDS + wn * WSTR + wkp,     src);
    async_ld_b128(wLDS + wn * WSTR + wkp + 8, src + 8);
  }

  // ---- stage token indices; OOB sequence -> token 0 (PAD row is all-zero)
  for (int r = tid; r < EROWS; r += 256) {
    int s = s0 - 2 + r;
    idxLDS[r] = (s >= 0 && s < SEQ) ? x[bb * SEQ + s] : 0;
  }
  __syncthreads();

  // ---- gather embedding rows, convert f32 -> f16 into LDS
  for (int i = tid; i < EROWS * (EMBD / 4); i += 256) {
    int r = i >> 5;           // EMBD/4 == 32 vec4 per row
    int c = (i & 31) << 2;
    const float4 v = *((const float4*)(emb + (size_t)idxLDS[r] * EMBD + c));
    v4h h;
    h[0] = (_Float16)v.x; h[1] = (_Float16)v.y;
    h[2] = (_Float16)v.z; h[3] = (_Float16)v.w;
    *((v4h*)(embLDS + r * ESTR + c)) = h;
  }

  v8f acc[2][4];
  #pragma unroll
  for (int ms = 0; ms < 2; ++ms)
    #pragma unroll
    for (int ns = 0; ns < 4; ++ns)
      #pragma unroll
      for (int e = 0; e < 8; ++e) acc[ms][ns][e] = 0.0f;

  wait_async0();     // W chunk 0 (this wave's slice) landed in LDS
  __syncthreads();   // gather + everyone's chunk-0 slices visible

  // K loop: global k = w*128 + e0 + kk, 20 steps of 32, double-buffered W chunk
  for (int kb = 0; kb < 20; ++kb) {
    const int w  = kb >> 2;
    const int e0 = (kb & 3) << 5;

    // ---- async-stage next chunk into the other buffer (no barrier needed yet)
    if (kb < 19) {
      const int kn  = kb + 1;
      const int wn2 = kn >> 2;
      const int en2 = (kn & 3) << 5;
      const _Float16* src = Wh + (size_t)wn * KTOT + wn2 * EMBD + en2 + wkp;
      _Float16* dst = wLDS + (kn & 1) * WBUFH + wn * WSTR + wkp;
      async_ld_b128(dst,     src);
      async_ld_b128(dst + 8, src + 8);
    }

    const _Float16* wbuf = wLDS + (kb & 1) * WBUFH;

    // ---- A fragments: 2 M-subtiles, window shift w, K-slice e0
    // a[i] = A[row][k], k = (i>>3)*16 + hi*8 + (i&7)  (ISA 16-bit A layout)
    v16h afr[2];
    #pragma unroll
    for (int ms = 0; ms < 2; ++ms) {
      const int row = (mw * 2 + ms) * 16 + ln + w;
      const _Float16* ap = embLDS + row * ESTR + e0 + hi * 8;
      v8h alo = *((const v8h*)ap);
      v8h ahi = *((const v8h*)(ap + 16));
      #pragma unroll
      for (int j = 0; j < 8; ++j) { afr[ms][j] = alo[j]; afr[ms][8 + j] = ahi[j]; }
    }

    // ---- 4 N-subtiles: B frag b[i] = B[k0 + (i>>3)*16 + hi*8 + (i&7)][n]
    #pragma unroll
    for (int ns = 0; ns < 4; ++ns) {
      const _Float16* bp = wbuf + (nw * 64 + ns * 16 + ln) * WSTR + hi * 8;
      v8h blo = *((const v8h*)bp);
      v8h bhi = *((const v8h*)(bp + 16));
      v16h bfr;
      #pragma unroll
      for (int j = 0; j < 8; ++j) { bfr[j] = blo[j]; bfr[8 + j] = bhi[j]; }
      acc[0][ns] = __builtin_amdgcn_wmma_f32_16x16x32_f16(
          false, afr[0], false, bfr, (short)0, acc[0][ns], false, false);
      acc[1][ns] = __builtin_amdgcn_wmma_f32_16x16x32_f16(
          false, afr[1], false, bfr, (short)0, acc[1][ns], false, false);
    }

    wait_async0();   // next chunk (this wave's slice) committed to LDS
    __syncthreads(); // all waves: done reading buf kb&1, next buffer visible
  }

  // ---- epilogue: +bias, exact GELU, store f32
  // D layout: value (m, n) with n-col = ln, m = v + 8*hi for accumulator elem v
  #pragma unroll
  for (int ms = 0; ms < 2; ++ms) {
    #pragma unroll
    for (int ns = 0; ns < 4; ++ns) {
      const int n  = nw * 64 + ns * 16 + ln;
      const float bv = bias[n];
      #pragma unroll
      for (int v = 0; v < 8; ++v) {
        const int m = v + 8 * hi;
        const int t = s0 + (mw * 2 + ms) * 16 + m;
        float y = acc[ms][ns][v] + bv;
        float g = 0.5f * y * (1.0f + erff(y * 0.70710678118654752f));
        out[((size_t)bb * SEQ + t) * OUTD + n] = g;
      }
    }
  }
}

extern "C" void kernel_launch(void* const* d_in, const int* in_sizes, int n_in,
                              void* d_out, int out_size, void* d_ws, size_t ws_size,
                              hipStream_t stream) {
  (void)in_sizes; (void)n_in; (void)out_size; (void)ws_size;
  const int*   x    = (const int*)d_in[0];
  const float* emb  = (const float*)d_in[1];
  const float* W    = (const float*)d_in[2];
  const float* bias = (const float*)d_in[3];
  float*       out  = (float*)d_out;
  _Float16*    Wh   = (_Float16*)d_ws;   // 128*640 f16 = 160 KB scratch

  const int wElems = OUTD * KTOT;
  w_to_f16<<<(wElems + 255) / 256, 256, 0, stream>>>(W, Wh, wElems);

  conv_encoder_wmma<<<BATCHN * (SEQ / MTILE), 256, 0, stream>>>(x, emb, Wh, bias, out);
}
